// OriginalHardContrastiveLossV6_8993661517970
// MI455X (gfx1250) — compile-verified, matrix operand order checked by reading, not run on hardware
//
#include <hip/hip_runtime.h>
#include <hip/hip_bf16.h>

typedef _Float16 v16h __attribute__((ext_vector_type(16)));
typedef _Float16 v8h  __attribute__((ext_vector_type(8)));
typedef float    v8f  __attribute__((ext_vector_type(8)));

#define TAU_F     0.2f
#define INV_TAU   5.0f
#define ALPHA_F   0.8f
#define EPS_F     1e-8f
#define DIM       128

// ---------------------------------------------------------------------------
// Kernel 1: build Mp bitmask + pos_partner map from the pairs list
// ---------------------------------------------------------------------------
__global__ void k_build_mp(const int* __restrict__ pairs, unsigned* __restrict__ mp,
                           int* __restrict__ pos_partner, int N, int P) {
    int p = blockIdx.x * blockDim.x + threadIdx.x;
    if (p >= P) return;
    int pi = pairs[2 * p + 0];
    int pj = pairs[2 * p + 1];
    int W = N >> 5;
    atomicOr(&mp[(size_t)pi * W + (pj >> 5)], 1u << (pj & 31));
    pos_partner[pi] = pj;   // mirrors jnp scatter (unique pi in this workload)
}

// ---------------------------------------------------------------------------
// Kernel 2: hard-negative partner via the posmat argmax logic
//   row_arg[v] = largest j (j!=v, !Mp[v][j]);  col_arg[v] = largest i likewise.
//   partner = col_max > row_max ? col_arg : row_arg  (values i*N+j, -1 if none)
// ---------------------------------------------------------------------------
__global__ void k_partner(const unsigned* __restrict__ mp, int* __restrict__ partner, int N) {
    int v = blockIdx.x * blockDim.x + threadIdx.x;
    if (v >= N) return;
    int W = N >> 5;
    long long rowmax = -1; int rowarg = 0;
    for (int w = W - 1; w >= 0; --w) {
        unsigned bits = mp[(size_t)v * W + w];
        if (w == (v >> 5)) bits |= 1u << (v & 31);   // exclude diagonal
        unsigned inv = ~bits;
        if (inv) {
            int j = (w << 5) + (31 - __clz(inv));
            rowarg = j; rowmax = (long long)v * N + j;
            break;
        }
    }
    long long colmax = -1; int colarg = 0;
    for (int i = N - 1; i >= 0; --i) {
        if (i == v) continue;
        if (!((mp[(size_t)i * W + (v >> 5)] >> (v & 31)) & 1u)) {
            colarg = i; colmax = (long long)i * N + v;
            break;
        }
    }
    partner[v] = (colmax > rowmax) ? colarg : rowarg;
}

// ---------------------------------------------------------------------------
// Kernel 3: hard_neg = 0.5*(e[v]+e[partner]); row-normalize; split into f16 hi/lo
// one block (128 threads) per row
// ---------------------------------------------------------------------------
__global__ __launch_bounds__(DIM) void k_prep(const float* __restrict__ emb,
                                              const int* __restrict__ partner,
                                              _Float16* __restrict__ Ahi,
                                              _Float16* __restrict__ Alo, int N) {
    int v = blockIdx.x;
    int t = threadIdx.x;
    __shared__ float red[DIM];
    float ev = emb[(size_t)v * DIM + t];
    int pr = partner[v];
    float hn = 0.5f * (ev + emb[(size_t)pr * DIM + t]);
    red[t] = hn * hn;
    __syncthreads();
    for (int off = DIM / 2; off > 0; off >>= 1) {
        if (t < off) red[t] += red[t + off];
        __syncthreads();
    }
    float nrm = sqrtf(red[0]);
    float h = hn / fmaxf(nrm, EPS_F);
    _Float16 hi = (_Float16)h;
    _Float16 lo = (_Float16)(h - (float)hi);
    Ahi[(size_t)v * DIM + t] = hi;
    Alo[(size_t)v * DIM + t] = lo;
}

// ---------------------------------------------------------------------------
// Kernel 4: WMMA GEMM  E = exp( Hn_hat @ Hn_hat^T / TAU ), zeroed on eye|Mp.
// 32x32 block tile, 4 waves, each wave one 16x16 WMMA tile.
// hi/lo split: c += hi*hi + hi*lo + lo*hi  (near-fp32 cosine accuracy)
// ---------------------------------------------------------------------------
__device__ inline v16h ld_frag(const _Float16* __restrict__ p) {
    // 16-bit A 16x32 layout: lane half selects K {0..7,16..23} vs {8..15,24..31}
    v8h a = *(const v8h*)(p);        // K = kb .. kb+7
    v8h b = *(const v8h*)(p + 16);   // K = kb+16 .. kb+23
    v16h r;
#pragma unroll
    for (int i = 0; i < 8; ++i) { r[i] = a[i]; r[i + 8] = b[i]; }
    return r;
}

__global__ __launch_bounds__(128) void k_gemm_exp(const _Float16* __restrict__ Ahi,
                                                  const _Float16* __restrict__ Alo,
                                                  const unsigned* __restrict__ mp,
                                                  float* __restrict__ E, int N) {
    int wave = threadIdx.x >> 5;
    int lane = threadIdx.x & 31;
    int R = blockIdx.y * 32 + (wave >> 1) * 16;   // output rows of this wave tile
    int C = blockIdx.x * 32 + (wave & 1) * 16;    // output cols of this wave tile
    int m   = lane & 15;
    int kh  = (lane >> 4) ? 8 : 0;                // lane-half K offset

    const _Float16* arH = Ahi + (size_t)(R + m) * DIM;
    const _Float16* arL = Alo + (size_t)(R + m) * DIM;
    const _Float16* brH = Ahi + (size_t)(C + m) * DIM;   // B frag = rows of A at col idx
    const _Float16* brL = Alo + (size_t)(C + m) * DIM;

    v8f c = {};
#pragma unroll
    for (int kc = 0; kc < DIM; kc += 32) {
        int kb = kc + kh;
        v16h ah = ld_frag(arH + kb);
        v16h bh = ld_frag(brH + kb);
        v16h al = ld_frag(arL + kb);
        v16h bl = ld_frag(brL + kb);
        c = __builtin_amdgcn_wmma_f32_16x16x32_f16(false, ah, false, bh, (short)0, c, false, false);
        c = __builtin_amdgcn_wmma_f32_16x16x32_f16(false, ah, false, bl, (short)0, c, false, false);
        c = __builtin_amdgcn_wmma_f32_16x16x32_f16(false, al, false, bh, (short)0, c, false, false);
    }

    // C/D layout: VGPR r -> M = r (lanes 0-15) / r+8 (lanes 16-31); N = lane&15
    int j = C + (lane & 15);
    int W = N >> 5;
#pragma unroll
    for (int r = 0; r < 8; ++r) {
        int i = R + r + ((lane >> 4) ? 8 : 0);
        bool msk = (i == j) || ((mp[(size_t)i * W + (j >> 5)] >> (j & 31)) & 1u);
        E[(size_t)i * N + j] = msk ? 0.0f : __expf(c[r] * INV_TAU);
    }
}

// ---------------------------------------------------------------------------
// Kernel 5: per-row quantile threshold (exact byte-radix select on fp32 bits,
// valid since all values >= 0) and thresholded sum. One 256-thread block/row.
// ---------------------------------------------------------------------------
#define RS_T 256
__device__ float kth_largest(const unsigned* sv, unsigned* hist, int* sel,
                             int N, int k, int t) {
    unsigned prefix = 0, msk = 0;
    int remaining = k;
    for (int shift = 24; shift >= 0; shift -= 8) {
        for (int b = t; b < 256; b += RS_T) hist[b] = 0;
        __syncthreads();
        for (int idx = t; idx < N; idx += RS_T) {
            unsigned x = sv[idx];
            if ((x & msk) == prefix) atomicAdd(&hist[(x >> shift) & 255u], 1u);
        }
        __syncthreads();
        if (t == 0) {
            int rem = remaining, b = 255;
            for (; b > 0; --b) {
                int cnt = (int)hist[b];
                if (rem - cnt <= 0) break;
                rem -= cnt;
            }
            sel[0] = b; sel[1] = rem;
        }
        __syncthreads();
        prefix |= ((unsigned)sel[0]) << shift;
        msk    |= 255u << shift;
        remaining = sel[1];
        __syncthreads();
    }
    return __uint_as_float(prefix);
}

__global__ __launch_bounds__(RS_T) void k_rowselect(const float* __restrict__ E,
                                                    const int* __restrict__ stage_p,
                                                    float* __restrict__ s, int N) {
    __shared__ unsigned sv[4096];
    __shared__ unsigned hist[256];
    __shared__ int sel[2];
    __shared__ float red[RS_T];
    int v = blockIdx.x;
    int t = threadIdx.x;
    int stage = stage_p[0];
    for (int idx = t; idx < N; idx += RS_T)
        sv[idx] = __float_as_uint(E[(size_t)v * N + idx]);
    __syncthreads();

    float ti = -1.0f;
    if (stage) {
        float fi = ALPHA_F * (float)(N - 1);
        int lo = (int)floorf(fi);
        float frac = fi - (float)lo;
        int k1 = N - lo;                       // k-th largest == ascending rank lo
        float Vlo = kth_largest(sv, hist, sel, N, k1, t);
        if (frac > 0.0f && lo + 1 <= N - 1) {
            float Vhi = kth_largest(sv, hist, sel, N, k1 - 1, t);
            ti = Vlo + frac * (Vhi - Vlo);
        } else {
            ti = Vlo;
        }
    }
    float acc = 0.0f;
    for (int idx = t; idx < N; idx += RS_T) {
        float x = __uint_as_float(sv[idx]);
        if (!stage || x >= ti) acc += x;
    }
    red[t] = acc;
    __syncthreads();
    for (int off = RS_T / 2; off > 0; off >>= 1) {
        if (t < off) red[t] += red[t + off];
        __syncthreads();
    }
    if (t == 0) s[v] = red[0];
}

// ---------------------------------------------------------------------------
// Kernel 6: per-pair loss. hard_pos rows rebuilt on the fly from pos_partner.
// one 128-thread block per pair; scalar tail on thread 0; atomic accumulate.
// ---------------------------------------------------------------------------
__global__ __launch_bounds__(DIM) void k_loss(const float* __restrict__ emb,
                                              const int* __restrict__ pairs,
                                              const int* __restrict__ pos_partner,
                                              const unsigned* __restrict__ mp,
                                              const float* __restrict__ E,
                                              const float* __restrict__ s,
                                              float* __restrict__ out, int N, int P) {
    __shared__ float r0[DIM], r1[DIM], r2[DIM];
    int p = blockIdx.x;
    int t = threadIdx.x;
    int pi = pairs[2 * p + 0];
    int pj = pairs[2 * p + 1];
    int qi = pos_partner[pi];
    int qj = pos_partner[pj];
    float ai = (qi >= 0) ? 1.5f * emb[(size_t)pi * DIM + t] - 0.5f * emb[(size_t)qi * DIM + t] : 0.0f;
    float aj = (qj >= 0) ? 1.5f * emb[(size_t)pj * DIM + t] - 0.5f * emb[(size_t)qj * DIM + t] : 0.0f;
    r0[t] = ai * aj; r1[t] = ai * ai; r2[t] = aj * aj;
    __syncthreads();
    for (int off = DIM / 2; off > 0; off >>= 1) {
        if (t < off) { r0[t] += r0[t + off]; r1[t] += r1[t + off]; r2[t] += r2[t + off]; }
        __syncthreads();
    }
    if (t == 0) {
        float cosv = r0[0] / (fmaxf(sqrtf(r1[0]), EPS_F) * fmaxf(sqrtf(r2[0]), EPS_F));
        float pos_ij = __expf(cosv * INV_TAU);
        float pos_ji;
        int W = N >> 5;
        bool mp_ji = (mp[(size_t)pj * W + (pi >> 5)] >> (pi & 31)) & 1u;
        if (mp_ji)            pos_ji = pos_ij;               // cosine is symmetric
        else if (pi == pj)    pos_ji = __expf(INV_TAU);      // diagonal: cos = 1
        else                  pos_ji = E[(size_t)pj * N + pi];
        float term = -logf(pos_ij / (pos_ij + s[pi])) - logf(pos_ji / (pos_ji + s[pj]));
        atomicAdd(out, term * (1.0f / (2.0f * (float)P)));
    }
}

// ---------------------------------------------------------------------------
// Host launch
// ---------------------------------------------------------------------------
extern "C" void kernel_launch(void* const* d_in, const int* in_sizes, int n_in,
                              void* d_out, int out_size, void* d_ws, size_t ws_size,
                              hipStream_t stream) {
    const float* emb    = (const float*)d_in[0];
    const int*   pairs  = (const int*)d_in[1];
    const int*   stage  = (const int*)d_in[2];
    float*       out    = (float*)d_out;

    const int N = in_sizes[0] / DIM;        // 4096
    const int P = in_sizes[1] / 2;          // 4096
    const int W = N >> 5;

    char* ws = (char*)d_ws;
    size_t off = 0;
    float*     E           = (float*)(ws + off);  off += (size_t)N * N * sizeof(float);
    _Float16*  Ahi         = (_Float16*)(ws + off); off += (size_t)N * DIM * sizeof(_Float16);
    _Float16*  Alo         = (_Float16*)(ws + off); off += (size_t)N * DIM * sizeof(_Float16);
    unsigned*  mp          = (unsigned*)(ws + off); off += (size_t)N * W * sizeof(unsigned);
    int*       partner     = (int*)(ws + off);      off += (size_t)N * sizeof(int);
    int*       pos_partner = (int*)(ws + off);      off += (size_t)N * sizeof(int);
    float*     srow        = (float*)(ws + off);    off += (size_t)N * sizeof(float);

    hipMemsetAsync(mp, 0, (size_t)N * W * sizeof(unsigned), stream);
    hipMemsetAsync(pos_partner, 0xFF, (size_t)N * sizeof(int), stream);  // -1
    hipMemsetAsync(out, 0, (size_t)out_size * sizeof(float), stream);

    k_build_mp<<<(P + 255) / 256, 256, 0, stream>>>(pairs, mp, pos_partner, N, P);
    k_partner <<<(N + 255) / 256, 256, 0, stream>>>(mp, partner, N);
    k_prep    <<<N, DIM, 0, stream>>>(emb, partner, Ahi, Alo, N);

    dim3 ggrid(N / 32, N / 32, 1);
    k_gemm_exp<<<ggrid, 128, 0, stream>>>(Ahi, Alo, mp, E, N);

    k_rowselect<<<N, RS_T, 0, stream>>>(E, stage, srow, N);
    k_loss     <<<P, DIM, 0, stream>>>(emb, pairs, pos_partner, mp, E, srow, out, N, P);
}
